// LatentBlockSeq_16252156248448
// MI455X (gfx1250) — matrix operand order repaired and checked by reading, not compiled, not run on hardware
//
#include <hip/hip_runtime.h>
#include <math.h>

// ---------------------------------------------------------------------------
// Types for CDNA5 WMMA (gfx1250, wave32)
// ---------------------------------------------------------------------------
typedef __bf16 bf16x8 __attribute__((ext_vector_type(8)));
typedef __bf16 v16bf  __attribute__((ext_vector_type(16)));
typedef float  v8f    __attribute__((ext_vector_type(8)));

union BF16Frag { v16bf v; bf16x8 h[2]; };

// Problem constants (from reference)
#define BB   4
#define SS   4096
#define DD   2048
#define LL   4
#define QHN  16
#define KVHN 4
#define HDN  128
#define CAP  1024
#define WIN  512
#define NTOK (BB * CAP)            // 4096 latent tokens total
#define EPSF 1.1920928955078125e-07f

// ---------------------------------------------------------------------------
// Router: rw = sigmoid(x @ router_w^T).  One wave per row.
// ---------------------------------------------------------------------------
__global__ __launch_bounds__(256) void router_kernel(
    const float* __restrict__ x, const float* __restrict__ rwgt,
    float* __restrict__ rw_ws, float* __restrict__ rw_out) {
  int row  = blockIdx.x * 8 + (threadIdx.x >> 5);  // 16384 rows
  int lane = threadIdx.x & 31;
  const float* xr = &x[(size_t)row * DD];
  float s = 0.f;
  for (int c = lane; c < DD; c += 32) s += xr[c] * rwgt[c];
  #pragma unroll
  for (int m = 1; m < 32; m <<= 1) s += __shfl_xor(s, m, 32);
  if (lane == 0) {
    float sg = 1.f / (1.f + __expf(-s));
    rw_ws[row] = sg;
    rw_out[row] = sg;
  }
}

// ---------------------------------------------------------------------------
// Stable rank (matches jax.lax.top_k tie-break: higher value first, then
// lower index).  selected = rank < CAP.
// ---------------------------------------------------------------------------
__global__ __launch_bounds__(256) void rank_kernel(
    const float* __restrict__ rw, int* __restrict__ selected, int* __restrict__ rankr) {
  int t = blockIdx.x * 256 + threadIdx.x;  // 0..16383
  int b = t >> 12, i = t & (SS - 1);
  const float* r = &rw[b * SS];
  float vi = r[i];
  int cnt = 0;
  for (int j = 0; j < SS; ++j) {
    float vj = r[j];
    cnt += (vj > vi) || (vj == vi && j < i);
  }
  selected[t] = (cnt < CAP) ? 1 : 0;
  rankr[t] = cnt;
}

// idx_sorted[m] = m-th smallest selected index; order[m] = its top-k rank.
// Faithful to the reference bug: top_w[m] = rw[b, order[m]]  (rank used as a
// sequence position).
__global__ __launch_bounds__(256) void build_kernel(
    const float* __restrict__ rw, const int* __restrict__ selected,
    const int* __restrict__ rankr, int* __restrict__ idxs, float* __restrict__ topw) {
  int t = blockIdx.x * 256 + threadIdx.x;
  int b = t >> 12, i = t & (SS - 1);
  if (!selected[t]) return;
  const int* sel = &selected[b * SS];
  int m = 0;
  for (int j = 0; j < i; ++j) m += sel[j];
  idxs[b * CAP + m] = i;
  topw[b * CAP + m] = rw[b * SS + rankr[t]];
}

__global__ __launch_bounds__(256) void dec_kernel(
    const int* __restrict__ selected, float* __restrict__ dec) {
  int t = blockIdx.x * 256 + threadIdx.x;
  dec[t] = selected[t] ? 1.0f : 0.0f;
}

__global__ __launch_bounds__(256) void gather_kernel(
    const float* __restrict__ x, const int* __restrict__ idxs, float* __restrict__ lat) {
  size_t e = (size_t)blockIdx.x * 256 + threadIdx.x;  // < NTOK*DD
  int tok = (int)(e >> 11), d = (int)(e & (DD - 1));
  int b = tok >> 10;
  lat[e] = x[((size_t)(b * SS + idxs[tok])) * DD + d];
}

// ---------------------------------------------------------------------------
// RMSNorm (fp32 in) -> bf16 out.  One 256-thread block per token.
// ---------------------------------------------------------------------------
__global__ __launch_bounds__(256) void rmsnorm_kernel(
    const float* __restrict__ X, const float* __restrict__ Wt, __bf16* __restrict__ Y) {
  int tok = blockIdx.x, tid = threadIdx.x;
  const float* xr = &X[(size_t)tok * DD];
  float v[8];
  float s = 0.f;
  #pragma unroll
  for (int k = 0; k < 8; ++k) { v[k] = xr[tid + k * 256]; s += v[k] * v[k]; }
  #pragma unroll
  for (int m = 1; m < 32; m <<= 1) s += __shfl_xor(s, m, 32);
  __shared__ float red[8];
  if ((tid & 31) == 0) red[tid >> 5] = s;
  __syncthreads();
  float tot = 0.f;
  #pragma unroll
  for (int wv = 0; wv < 8; ++wv) tot += red[wv];
  float rms = rsqrtf(tot * (1.0f / (float)DD) + EPSF);
  #pragma unroll
  for (int k = 0; k < 8; ++k) {
    int c = tid + k * 256;
    Y[(size_t)tok * DD + c] = (__bf16)(v[k] * rms * Wt[c]);
  }
}

// ---------------------------------------------------------------------------
// WMMA GEMM:  C[M,N] = A[M,K] (bf16, row-major) @ W[N,K]^T (fp32, row-major).
// Block tile 64x128, 8 waves each computing 32x32 via 2x2 v_wmma_f32_16x16x32_bf16.
// K-stage = 64 (two 32-wide WMMA steps per LDS stage -> 8 WMMAs per wave
// between barriers, halving sync overhead vs a 32-wide stage).
// Weights converted fp32->bf16 in the LDS tile loader.  Double-buffered LDS.
// RES=0: bf16 output.  RES=1: fp32 residual accumulate into OutF.
// ---------------------------------------------------------------------------
#define KSTEP 64
#define LSTR  72   // padded LDS row stride (bf16 elems); 144B rows, 16B aligned
__global__ __launch_bounds__(256) void gemm_wmma(
    const __bf16* __restrict__ A, const float* __restrict__ W,
    __bf16* __restrict__ OutB, float* __restrict__ OutF, int N, int K, int RES) {
  __shared__ __bf16 As[2 * 64 * LSTR];    // 18.4 KB
  __shared__ __bf16 Bs[2 * 128 * LSTR];   // 36.9 KB
  const int tid = threadIdx.x, lane = tid & 31, wid = tid >> 5;
  const int wm = wid >> 2, wn = wid & 3;        // 2x4 wave grid
  const int hs = lane >> 4, lr = lane & 15;
  const int mBase = blockIdx.x * 64, nBase = blockIdx.y * 128;
  const int arow = tid >> 2, acol = (tid & 3) * 16;  // A: 64x64 tile, 16 bf16/thread
  const int brow = tid >> 1, bcol = (tid & 1) * 32;  // W: 128x64 tile, 32 f32/thread

  v8f acc[2][2];
  {
    v8f z = {};
    acc[0][0] = z; acc[0][1] = z; acc[1][0] = z; acc[1][1] = z;
  }

  bf16x8 ra[2];
  float4 rw4[8];
  auto loadG = [&](int k0) {
    const bf16x8* ap = (const bf16x8*)&A[(size_t)(mBase + arow) * K + k0 + acol];
    ra[0] = ap[0]; ra[1] = ap[1];
    const float4* wp = (const float4*)&W[(size_t)(nBase + brow) * K + k0 + bcol];
    #pragma unroll
    for (int q = 0; q < 8; ++q) rw4[q] = wp[q];
  };
  auto storeL = [&](int buf) {
    __bf16* ad = &As[buf * 64 * LSTR + arow * LSTR + acol];
    *(bf16x8*)(ad)     = ra[0];
    *(bf16x8*)(ad + 8) = ra[1];
    __bf16* bd = &Bs[buf * 128 * LSTR + brow * LSTR + bcol];
    #pragma unroll
    for (int q = 0; q < 8; ++q) {
      bd[q * 4 + 0] = (__bf16)rw4[q].x;
      bd[q * 4 + 1] = (__bf16)rw4[q].y;
      bd[q * 4 + 2] = (__bf16)rw4[q].z;
      bd[q * 4 + 3] = (__bf16)rw4[q].w;
    }
  };

  loadG(0);
  storeL(0);
  __syncthreads();
  const int KT = K / KSTEP;
  for (int kt = 0; kt < KT; ++kt) {
    int cur = kt & 1;
    if (kt + 1 < KT) loadG((kt + 1) * KSTEP);
    if (kt + 2 < KT)  // gfx1250 global_prefetch_b8 for the stage after next
      __builtin_prefetch(&W[(size_t)(nBase + brow) * K + (kt + 2) * KSTEP + bcol], 0, 3);

    #pragma unroll
    for (int kk = 0; kk < KSTEP; kk += 32) {
      v16bf afr[2], bfr[2];
      #pragma unroll
      for (int i = 0; i < 2; ++i) {  // A frag: lane<16 K{0..7,16..23}, lane>=16 +8
        BF16Frag f;
        const __bf16* ab = &As[cur * 64 * LSTR + (wm * 32 + 16 * i + lr) * LSTR + kk];
        f.h[0] = *(const bf16x8*)(ab + hs * 8);
        f.h[1] = *(const bf16x8*)(ab + 16 + hs * 8);
        afr[i] = f.v;
      }
      #pragma unroll
      for (int j = 0; j < 2; ++j) {  // B frag: lane<16 K{0..7,16..23}, lane>=16 +8
        BF16Frag f;
        const __bf16* bb =
            &Bs[cur * 128 * LSTR + (wn * 32 + 16 * j + lr) * LSTR + kk];
        f.h[0] = *(const bf16x8*)(bb + hs * 8);
        f.h[1] = *(const bf16x8*)(bb + 16 + hs * 8);
        bfr[j] = f.v;
      }
      #pragma unroll
      for (int i = 0; i < 2; ++i)
        #pragma unroll
        for (int j = 0; j < 2; ++j)
          acc[i][j] = __builtin_amdgcn_wmma_f32_16x16x32_bf16(
              false, afr[i], false, bfr[j], (short)0, acc[i][j], false, false);
    }
    __syncthreads();
    if (kt + 1 < KT) { storeL((kt + 1) & 1); __syncthreads(); }
  }

  // Epilogue: C layout — lane holds column lr, rows r + 8*hs.
  #pragma unroll
  for (int i = 0; i < 2; ++i) {
    int row0 = mBase + wm * 32 + 16 * i + 8 * hs;
    #pragma unroll
    for (int j = 0; j < 2; ++j) {
      int col = nBase + wn * 32 + 16 * j + lr;
      #pragma unroll
      for (int r = 0; r < 8; ++r) {
        size_t idx = (size_t)(row0 + r) * N + col;
        if (RES) OutF[idx] += acc[i][j][r];
        else     OutB[idx] = (__bf16)acc[i][j][r];
      }
    }
  }
}

// ---------------------------------------------------------------------------
// Flash-style GQA attention (causal sliding window 512 + ALiBi), all WMMA.
// Grid: x = B*QH (64), y = CAP/128 (8).  Block = 256 threads (8 waves).
// Wave w owns q rows [q0 + 16w, +16).  Keys iterate in 32-wide tiles over the
// union window [max(0,q0-512), q0+128); per-element mask handles the rest.
// QKV buffer: [token, 3072] bf16 = [q(2048) | k(512) | v(512)].
// ---------------------------------------------------------------------------
__global__ __launch_bounds__(256) void attn_kernel(
    const __bf16* __restrict__ QKV, __bf16* __restrict__ Att) {
  const int bh = blockIdx.x;
  const int b = bh >> 4, h = bh & 15;
  const int kvh = h >> 2;
  const int q0 = blockIdx.y * 128;
  const int tid = threadIdx.x, w = tid >> 5, lane = tid & 31;
  const int hs = lane >> 4, lr = lane & 15;
  const float slope = exp2f(-0.5f * (float)(h + 1));  // 2^(-(h+1)*8/16)
  const float scale = 0.08838834764831845f;           // 1/sqrt(128)

  __shared__ __bf16 Qs[128 * 128];   // 32KB  [qrow][hd]
  __shared__ __bf16 Ks[32 * 128];    //  8KB  [key][hd]
  __shared__ __bf16 Vt[128 * 32];    //  8KB  [hd][key] (transposed)
  __shared__ __bf16 Ps[8 * 16 * 32]; //  8KB  per-wave P scratch

  {  // cooperative Q load (once)
    int qrow = tid >> 1, hd0 = (tid & 1) * 64;
    const __bf16* src = &QKV[(size_t)(b * CAP + q0 + qrow) * 3072 + h * HDN + hd0];
    #pragma unroll
    for (int c = 0; c < 8; ++c)
      *(bf16x8*)&Qs[qrow * 128 + hd0 + c * 8] = *(const bf16x8*)&src[c * 8];
  }

  v8f o[8];
  {
    v8f z = {};
    #pragma unroll
    for (int c = 0; c < 8; ++c) o[c] = z;
  }
  float m_st[8], l_st[8];
  #pragma unroll
  for (int r = 0; r < 8; ++r) { m_st[r] = -INFINITY; l_st[r] = 0.f; }

  int jstart = q0 - WIN; if (jstart < 0) jstart = 0;
  const int jend = q0 + 128;

  for (int kb = jstart; kb < jend; kb += 32) {
    __syncthreads();  // previous tile fully consumed (also covers Qs at t=0)
    {  // cooperative K/V tile load; V stored transposed for the B fragment
      int key = tid >> 3, hd0 = (tid & 7) * 16;
      size_t trow = (size_t)(b * CAP + kb + key) * 3072;
      const __bf16* ksrc = &QKV[trow + 2048 + kvh * HDN + hd0];
      *(bf16x8*)&Ks[key * 128 + hd0]     = *(const bf16x8*)&ksrc[0];
      *(bf16x8*)&Ks[key * 128 + hd0 + 8] = *(const bf16x8*)&ksrc[8];
      const __bf16* vsrc = &QKV[trow + 2560 + kvh * HDN + hd0];
      bf16x8 v0 = *(const bf16x8*)&vsrc[0];
      bf16x8 v1 = *(const bf16x8*)&vsrc[8];
      #pragma unroll
      for (int e = 0; e < 8; ++e) {
        Vt[(hd0 + e) * 32 + key]     = v0[e];
        Vt[(hd0 + 8 + e) * 32 + key] = v1[e];
      }
    }
    __syncthreads();

    // S = Q K^T : 16x32 scores via 2 accumulators, K-dim = HD (4 steps of 32)
    v8f s0 = {}, s1 = {};
    #pragma unroll
    for (int kc = 0; kc < HDN; kc += 32) {
      BF16Frag af;
      const __bf16* ab = &Qs[(w * 16 + lr) * 128 + kc];
      af.h[0] = *(const bf16x8*)(ab + hs * 8);
      af.h[1] = *(const bf16x8*)(ab + 16 + hs * 8);
      BF16Frag b0, b1;
      const __bf16* k0p = &Ks[(0 * 16 + lr) * 128 + kc];
      const __bf16* k1p = &Ks[(1 * 16 + lr) * 128 + kc];
      b0.h[0] = *(const bf16x8*)(k0p + hs * 8); b0.h[1] = *(const bf16x8*)(k0p + 16 + hs * 8);
      b1.h[0] = *(const bf16x8*)(k1p + hs * 8); b1.h[1] = *(const bf16x8*)(k1p + 16 + hs * 8);
      s0 = __builtin_amdgcn_wmma_f32_16x16x32_bf16(false, af.v, false, b0.v, (short)0, s0, false, false);
      s1 = __builtin_amdgcn_wmma_f32_16x16x32_bf16(false, af.v, false, b1.v, (short)0, s1, false, false);
    }

    // Online softmax; row reductions via shfl_xor within 16-lane halves
    #pragma unroll
    for (int r = 0; r < 8; ++r) {
      int m_abs = q0 + w * 16 + r + 8 * hs;
      int d0 = m_abs - (kb + lr);
      int d1 = m_abs - (kb + 16 + lr);
      float sv0 = (d0 >= 0 && d0 <= WIN) ? s0[r] * scale - slope * (float)d0 : -INFINITY;
      float sv1 = (d1 >= 0 && d1 <= WIN) ? s1[r] * scale - slope * (float)d1 : -INFINITY;
      float tm = fmaxf(sv0, sv1);
      #pragma unroll
      for (int m = 1; m < 16; m <<= 1) tm = fmaxf(tm, __shfl_xor(tm, m, 32));
      float mn = fmaxf(m_st[r], tm);
      float corr = (m_st[r] > -INFINITY) ? __expf(m_st[r] - mn) : 0.f;
      float p0 = (sv0 > -INFINITY) ? __expf(sv0 - mn) : 0.f;
      float p1 = (sv1 > -INFINITY) ? __expf(sv1 - mn) : 0.f;
      float ts = p0 + p1;
      #pragma unroll
      for (int m = 1; m < 16; m <<= 1) ts += __shfl_xor(ts, m, 32);
      l_st[r] = l_st[r] * corr + ts;
      m_st[r] = mn;
      #pragma unroll
      for (int c = 0; c < 8; ++c) o[c][r] *= corr;
      int prow = (w * 16 + r + 8 * hs) * 32;
      Ps[prow + lr]      = (__bf16)p0;
      Ps[prow + 16 + lr] = (__bf16)p1;
    }
    asm volatile("s_wait_dscnt 0" ::: "memory");  // P visible to own wave's DS loads

    // O += P V : A = P (16x32), B = V^T tiles (keys x 16 hd), 8 WMMA
    BF16Frag ap;
    const __bf16* pb = &Ps[(w * 16 + lr) * 32];
    ap.h[0] = *(const bf16x8*)(pb + hs * 8);
    ap.h[1] = *(const bf16x8*)(pb + 16 + hs * 8);
    #pragma unroll
    for (int c = 0; c < 8; ++c) {
      BF16Frag bv;
      const __bf16* vb = &Vt[(c * 16 + lr) * 32];
      bv.h[0] = *(const bf16x8*)(vb + hs * 8);
      bv.h[1] = *(const bf16x8*)(vb + 16 + hs * 8);
      o[c] = __builtin_amdgcn_wmma_f32_16x16x32_bf16(false, ap.v, false, bv.v, (short)0, o[c], false, false);
    }
  }

  #pragma unroll
  for (int r = 0; r < 8; ++r) {
    int tok = b * CAP + q0 + w * 16 + r + 8 * hs;
    float inv = 1.0f / l_st[r];
    #pragma unroll
    for (int c = 0; c < 8; ++c)
      Att[(size_t)tok * DD + h * HDN + c * 16 + lr] = (__bf16)(o[c][r] * inv);
  }
}

// ---------------------------------------------------------------------------
// SwiGLU: g = silu(h[:,2048:]) * h[:,:2048]   (bf16 in/out, fp32 math)
// ---------------------------------------------------------------------------
__global__ __launch_bounds__(256) void swiglu_kernel(
    const __bf16* __restrict__ H, __bf16* __restrict__ G) {
  size_t e = (size_t)blockIdx.x * 256 + threadIdx.x;  // NTOK*DD
  int tok = (int)(e >> 11), d = (int)(e & (DD - 1));
  float a = (float)H[(size_t)tok * (2 * DD) + d];
  float g = (float)H[(size_t)tok * (2 * DD) + DD + d];
  float sg = g / (1.f + __expf(-g));
  G[e] = (__bf16)(sg * a);
}

__global__ __launch_bounds__(256) void zero_kernel(float* __restrict__ p, size_t n) {
  size_t e = (size_t)blockIdx.x * 256 + threadIdx.x;
  if (e < n) p[e] = 0.f;
}

__global__ __launch_bounds__(256) void scatter_kernel(
    const float* __restrict__ lat, const float* __restrict__ topw,
    const int* __restrict__ idxs, float* __restrict__ pred) {
  size_t e = (size_t)blockIdx.x * 256 + threadIdx.x;  // NTOK*DD
  int tok = (int)(e >> 11), d = (int)(e & (DD - 1));
  int b = tok >> 10;
  pred[((size_t)(b * SS + idxs[tok])) * DD + d] = lat[e] * topw[tok];
}

// ---------------------------------------------------------------------------
// Launch
// ---------------------------------------------------------------------------
extern "C" void kernel_launch(void* const* d_in, const int* in_sizes, int n_in,
                              void* d_out, int out_size, void* d_ws, size_t ws_size,
                              hipStream_t stream) {
  (void)in_sizes; (void)n_in; (void)out_size; (void)ws_size;
  const float* x        = (const float*)d_in[0];
  const float* router_w = (const float*)d_in[1];
  const float* norm1_w  = (const float*)d_in[2];
  const float* norm2_w  = (const float*)d_in[3];
  const float* qkv_w    = (const float*)d_in[4];
  const float* proj_w   = (const float*)d_in[5];
  const float* fc1_w    = (const float*)d_in[6];
  const float* fc2_w    = (const float*)d_in[7];

  float* pred   = (float*)d_out;                       // [B,S,D]
  float* rw_out = pred + (size_t)BB * SS * DD;         // [B,S,1]
  float* dec    = rw_out + (size_t)BB * SS;            // [B,S,1]

  char* p = (char*)d_ws;
  auto alloc = [&](size_t bytes) { void* r = p; p += (bytes + 255) & ~(size_t)255; return r; };
  float*  rw_ws    = (float*)alloc((size_t)BB * SS * 4);
  int*    selected = (int*)  alloc((size_t)BB * SS * 4);
  int*    rankr    = (int*)  alloc((size_t)BB * SS * 4);
  int*    idxs     = (int*)  alloc((size_t)NTOK * 4);
  float*  topw     = (float*)alloc((size_t)NTOK * 4);
  float*  lat      = (float*)alloc((size_t)NTOK * DD * 4);
  __bf16* xn       = (__bf16*)alloc((size_t)NTOK * DD * 2);
  __bf16* qkvb     = (__bf16*)alloc((size_t)NTOK * 3072 * 2);
  __bf16* attb     = (__bf16*)alloc((size_t)NTOK * DD * 2);
  __bf16* hb       = (__bf16*)alloc((size_t)NTOK * 2 * DD * 2);
  __bf16* gb       = (__bf16*)alloc((size_t)NTOK * DD * 2);

  const size_t predN = (size_t)BB * SS * DD;

  router_kernel<<<(BB * SS) / 8, 256, 0, stream>>>(x, router_w, rw_ws, rw_out);
  zero_kernel<<<(unsigned)((predN + 255) / 256), 256, 0, stream>>>(pred, predN);
  rank_kernel<<<(BB * SS) / 256, 256, 0, stream>>>(rw_ws, selected, rankr);
  build_kernel<<<(BB * SS) / 256, 256, 0, stream>>>(rw_ws, selected, rankr, idxs, topw);
  dec_kernel<<<(BB * SS) / 256, 256, 0, stream>>>(selected, dec);
  gather_kernel<<<(NTOK * DD) / 256, 256, 0, stream>>>(x, idxs, lat);

  for (int l = 0; l < LL; ++l) {
    rmsnorm_kernel<<<NTOK, 256, 0, stream>>>(lat, norm1_w + l * DD, xn);
    gemm_wmma<<<dim3(NTOK / 64, 3072 / 128), 256, 0, stream>>>(
        xn, qkv_w + (size_t)l * 3072 * DD, qkvb, nullptr, 3072, DD, 0);
    attn_kernel<<<dim3(BB * QHN, CAP / 128), 256, 0, stream>>>(qkvb, attb);
    gemm_wmma<<<dim3(NTOK / 64, DD / 128), 256, 0, stream>>>(
        attb, proj_w + (size_t)l * DD * DD, nullptr, lat, DD, DD, 1);
    rmsnorm_kernel<<<NTOK, 256, 0, stream>>>(lat, norm2_w + l * DD, xn);
    gemm_wmma<<<dim3(NTOK / 64, (2 * DD) / 128), 256, 0, stream>>>(
        xn, fc1_w + (size_t)l * 2 * DD * DD, hb, nullptr, 2 * DD, DD, 0);
    swiglu_kernel<<<(NTOK * DD) / 256, 256, 0, stream>>>(hb, gb);
    gemm_wmma<<<dim3(NTOK / 64, DD / 128), 256, 0, stream>>>(
        gb, fc2_w + (size_t)l * DD * DD, nullptr, lat, DD, DD, 1);
  }

  scatter_kernel<<<(NTOK * DD) / 256, 256, 0, stream>>>(lat, topw, idxs, pred);
}
